// LocalReconstructor_51539607552757
// MI455X (gfx1250) — matrix operand order repaired and checked by reading, not compiled, not run on hardware
//
#include <hip/hip_runtime.h>
#include <hip/hip_bf16.h>

// ---------------------------------------------------------------------------
// LocalReconstructor for MI455X (gfx1250, wave32, WMMA).
//
// Dominant work: per-step gates GEMM (256 x 4096 x 2048) -> bf16 WMMA
// (v_wmma_f32_16x16x32_bf16, f32 accumulate), register-blocked 16x64 per wave
// (1.25 fragment loads per wmma) and software-pipelined one k-step ahead so
// L2 latency overlaps wmma execution. Weights pre-packed once per launch into
// WMMA B-fragment layout; activations (h|ctx) kept in a packed A-fragment
// buffer between steps. Everything hot is L2-resident (192 MB); the 28-step
// serial chain is the latency limiter, so each GEMM uses the highest-K
// f32-safe WMMA path (bf16, K=32).
// ---------------------------------------------------------------------------

#define B_   256
#define T_CAP 64
#define T_FEAT 28
#define DEC  1024
#define HID  1024
#define ATTN 128
#define G4   4096          // 4*HID
#define KCOMB 2048         // [h(1024) | ctx(1024)]
#define KT_COMB 64         // KCOMB/32
#define KT_H 32            // per-1024 K tiles
#define MT   16            // B_/16
#define NT_G 256           // G4/16
#define NT_A 8             // ATTN/16

typedef __attribute__((ext_vector_type(16))) __bf16 v16bf;
typedef __attribute__((ext_vector_type(8)))  float  v8f;

__device__ __forceinline__ unsigned short f2bf(float f) {
    union { float f; unsigned u; } v; v.f = f;
    unsigned u = v.u;
    u += 0x7FFFu + ((u >> 16) & 1u);       // round-to-nearest-even
    return (unsigned short)(u >> 16);
}

union FragA { v16bf v; unsigned short s[16]; };

#define WMMA_BF16(A, Bv, C) \
    __builtin_amdgcn_wmma_f32_16x16x32_bf16(false, (A), false, (Bv), (short)0, (C), false, false)

// ---------------------------------------------------------------------------
// Weight packing into WMMA B-fragment layout.
// B frag (16-bit, 32x16): lane = 16*half + n ; element jj <-> K = kBase+8*half+jj (jj<8), kBase+16+8*half+(jj-8) (jj>=8)
// dst layout: [nTile][kTile][lane][16] bf16  -> 32B/lane coalesced frag loads.
// ---------------------------------------------------------------------------
__global__ void pack_b_kernel(const float* __restrict__ W, unsigned short* __restrict__ dst,
                              int nTiles, int kTiles, int K) {
    int idx = blockIdx.x * blockDim.x + threadIdx.x;       // (nt*kTiles+kt)*32+lane
    if (idx >= nTiles * kTiles * 32) return;
    int lane = idx & 31;
    int kt   = (idx >> 5) % kTiles;
    int nt   = (idx >> 5) / kTiles;
    int half = lane >> 4, n = lane & 15;
    const float* src = W + (size_t)(nt * 16 + n) * K + kt * 32 + 8 * half;
    unsigned short* d = dst + (size_t)idx * 16;
#pragma unroll
    for (int j = 0; j < 8; ++j) d[j] = f2bf(src[j]);
#pragma unroll
    for (int j = 0; j < 8; ++j) d[8 + j] = f2bf(src[16 + j]);
}

// Combined [W_hh ; W_ih] along K: kTile 0..31 -> W_hh, 32..63 -> W_ih.
__global__ void pack_comb_kernel(const float* __restrict__ Whh, const float* __restrict__ Wih,
                                 unsigned short* __restrict__ dst) {
    int idx = blockIdx.x * blockDim.x + threadIdx.x;       // (nt*64+kt)*32+lane
    if (idx >= NT_G * KT_COMB * 32) return;
    int lane = idx & 31;
    int kt   = (idx >> 5) & 63;
    int nt   = idx >> 11;
    int half = lane >> 4, n = lane & 15;
    const float* W = (kt < 32) ? Whh : Wih;
    const float* src = W + (size_t)(nt * 16 + n) * HID + (kt & 31) * 32 + 8 * half;
    unsigned short* d = dst + (size_t)idx * 16;
#pragma unroll
    for (int j = 0; j < 8; ++j) d[j] = f2bf(src[j]);
#pragma unroll
    for (int j = 0; j < 8; ++j) d[8 + j] = f2bf(src[16 + j]);
}

// ---------------------------------------------------------------------------
// Pack a 256x1024 f32 activation into A-fragment layout at kTile offset.
// A frag (16-bit, 16x32): lane L holds row M=L&15; half = L>>4:
//   elems 0..7  <-> K = kBase + 8*half + j
//   elems 8..15 <-> K = kBase + 16 + 8*half + (j-8)
// aFrags layout: [mTile][kTileGlobal(64)][lane][16] bf16.
// ---------------------------------------------------------------------------
__global__ void pack_a_kernel(const float* __restrict__ src, unsigned short* __restrict__ aFrags,
                              int ktOff) {
    int idx = blockIdx.x * blockDim.x + threadIdx.x;       // (mTile*32+kt)*32+lane
    if (idx >= MT * KT_H * 32) return;
    int lane = idx & 31;
    int kt   = (idx >> 5) & 31;
    int mTile = idx >> 10;
    int half = lane >> 4;
    int m = mTile * 16 + (lane & 15);
    const float* row = src + (size_t)m * 1024 + kt * 32 + 8 * half;
    unsigned short* d = aFrags + (((size_t)mTile * KT_COMB + ktOff + kt) * 32 + lane) * 16;
#pragma unroll
    for (int j = 0; j < 8; ++j) d[j]     = f2bf(row[j]);
#pragma unroll
    for (int j = 0; j < 8; ++j) d[8 + j] = f2bf(row[16 + j]);
}

// ---------------------------------------------------------------------------
// One-time: Uv[b*64+t][a] = feats(b,t,:) . U_attn[a,:] + b_attn[a]
// Each wave computes a full 16x128 row-stripe (all 8 ATTN n-tiles) so the
// f32->bf16 A conversion is amortized over 8 wmmas per k-step.
// 1024 waves, 32 k-steps x 8 wmma each.
// ---------------------------------------------------------------------------
__global__ void uv_gemm_kernel(const float* __restrict__ dh, const unsigned short* __restrict__ uFrags,
                               const float* __restrict__ b_attn, float* __restrict__ Uv) {
    int wave = (blockIdx.x * blockDim.x + threadIdx.x) >> 5;   // 0..1023 = mTile
    int lane = threadIdx.x & 31;
    int mTile = wave;
    int half = lane >> 4;
    int mrow = mTile * 16 + (lane & 15);                   // R = b*64 + t
    int b = mrow >> 6, t = mrow & 63;
    const float* arow = dh + ((size_t)t * B_ + b) * DEC;
    const v16bf* U = (const v16bf*)uFrags;                 // frag (nt,k): U[(nt*KT_H+k)*32+lane]
    v8f acc[NT_A];
#pragma unroll
    for (int nt = 0; nt < NT_A; ++nt) acc[nt] = (v8f){};
    for (int k = 0; k < KT_H; ++k) {
        int base = k * 32 + 8 * half;
        FragA a;
#pragma unroll
        for (int j = 0; j < 8; ++j) a.s[j]     = f2bf(arow[base + j]);
#pragma unroll
        for (int j = 0; j < 8; ++j) a.s[8 + j] = f2bf(arow[base + 16 + j]);
#pragma unroll
        for (int nt = 0; nt < NT_A; ++nt)
            acc[nt] = WMMA_BF16(a.v, U[((size_t)nt * KT_H + k) * 32 + lane], acc[nt]);
    }
#pragma unroll
    for (int nt = 0; nt < NT_A; ++nt) {
        int n = nt * 16 + (lane & 15);
        float bias = b_attn[n];
#pragma unroll
        for (int r = 0; r < 8; ++r)
            Uv[(size_t)(mTile * 16 + r + 8 * half) * ATTN + n] = acc[nt][r] + bias;
    }
}

// ---------------------------------------------------------------------------
// Per step: Wh = h @ W_attn^T  (256x128, K=1024). Uses h region (kTiles 0..31)
// of the persistent A-fragment buffer. 16x8 = 128 waves (latency-optimal for
// this tiny GEMM), 32 wmma each.
// ---------------------------------------------------------------------------
__global__ void wh_gemm_kernel(const unsigned short* __restrict__ aFrags,
                               const unsigned short* __restrict__ wFrags,
                               float* __restrict__ Wh) {
    int wave = (blockIdx.x * blockDim.x + threadIdx.x) >> 5;
    int lane = threadIdx.x & 31;
    int mTile = wave >> 3, nTile = wave & 7;
    const v16bf* A = (const v16bf*)aFrags;
    const v16bf* Bv = (const v16bf*)wFrags;
    v8f acc = {};
    for (int k = 0; k < KT_H; ++k)
        acc = WMMA_BF16(A[((size_t)mTile * KT_COMB + k) * 32 + lane],
                        Bv[((size_t)nTile * KT_H + k) * 32 + lane], acc);
    int n = nTile * 16 + (lane & 15), half = lane >> 4;
#pragma unroll
    for (int r = 0; r < 8; ++r)
        Wh[(size_t)(mTile * 16 + r + 8 * half) * ATTN + n] = acc[r];
}

// ---------------------------------------------------------------------------
// Per step: energies -> masked softmax -> ctx. One block per batch row b.
// feats read straight from decoder_hiddens out of L2 (float4 coalesced).
// ---------------------------------------------------------------------------
__global__ void attention_ctx_kernel(const float* __restrict__ Wh, const float* __restrict__ Uv,
                                     const float* __restrict__ w_attn,
                                     const int* __restrict__ captions,
                                     const float* __restrict__ dh,
                                     float* __restrict__ ctx) {
    int b = blockIdx.x;
    int tid = threadIdx.x;                                  // 256 threads = 8 waves
    int wave = tid >> 5, lane = tid & 31;
    __shared__ float e[T_CAP];
    __shared__ float wn[T_CAP];

    // energies: wave w handles t = w, w+8, ... ; 128-wide dot over ATTN
    for (int t = wave; t < T_CAP; t += 8) {
        const float* uvp = Uv + ((size_t)b * T_CAP + t) * ATTN;
        const float* whp = Wh + (size_t)b * ATTN;
        float p = 0.f;
#pragma unroll
        for (int q = 0; q < 4; ++q) {
            int a = lane + 32 * q;
            p += tanhf(whp[a] + uvp[a]) * w_attn[a];
        }
#pragma unroll
        for (int off = 16; off; off >>= 1) p += __shfl_xor(p, off, 32);
        if (lane == 0) e[t] = p;
    }
    __syncthreads();

    if (tid == 0) {                                         // 64-elem serial softmax (cheap)
        float mx = -3.4e38f;
        for (int t = 0; t < T_CAP; ++t) {
            int cap = captions[t * B_ + b];
            float v = (cap != 0 && cap != 2) ? e[t] : -3.4e38f;
            e[t] = v;
            mx = fmaxf(mx, v);
        }
        float s = 0.f;
        for (int t = 0; t < T_CAP; ++t) { float w = __expf(e[t] - mx); wn[t] = w; s += w; }
        float inv = 1.f / s;
        for (int t = 0; t < T_CAP; ++t) wn[t] *= inv;
    }
    __syncthreads();

    // ctx[b,d] = sum_t wn[t] * feats[b,t,d]; each thread owns 4 d's (float4)
    float4 acc = make_float4(0.f, 0.f, 0.f, 0.f);
    for (int t = 0; t < T_CAP; ++t) {
        float w = wn[t];
        const float4 f = *(const float4*)(dh + ((size_t)t * B_ + b) * DEC + tid * 4);
        acc.x += w * f.x; acc.y += w * f.y; acc.z += w * f.z; acc.w += w * f.w;
    }
    *(float4*)(ctx + (size_t)b * DEC + tid * 4) = acc;
}

// ---------------------------------------------------------------------------
// Per step (dominant): gates = [h|ctx] @ [W_hh;W_ih]^T + b_ih + b_hh
// 256x4096, K=2048. Register-blocked 16x64 per wave: 1024 waves, each
// 64 k-steps x (1 A-frag + 4 B-frags -> 4 wmma). Fragments for k+1 are
// loaded while the wmmas for k execute (software pipeline / double buffer),
// so the loadcnt wait lands where L2 latency is already hidden.
// ---------------------------------------------------------------------------
__global__ void gates_gemm_kernel(const unsigned short* __restrict__ aFrags,
                                  const unsigned short* __restrict__ bFrags,
                                  const float* __restrict__ b_ih, const float* __restrict__ b_hh,
                                  float* __restrict__ gates) {
    int wave = (blockIdx.x * blockDim.x + threadIdx.x) >> 5;
    int lane = threadIdx.x & 31;
    int mTile = wave >> 6;                                  // 0..15
    int nGrp  = wave & 63;                                  // 0..63 -> 4 nTiles each
    const v16bf* A  = (const v16bf*)aFrags;
    const v16bf* Bc = (const v16bf*)bFrags;
    size_t aBase  = (size_t)mTile * KT_COMB * 32 + lane;
    size_t bBase0 = ((size_t)(nGrp * 4 + 0) * KT_COMB) * 32 + lane;
    size_t bBase1 = ((size_t)(nGrp * 4 + 1) * KT_COMB) * 32 + lane;
    size_t bBase2 = ((size_t)(nGrp * 4 + 2) * KT_COMB) * 32 + lane;
    size_t bBase3 = ((size_t)(nGrp * 4 + 3) * KT_COMB) * 32 + lane;
    v8f acc0 = {}, acc1 = {}, acc2 = {}, acc3 = {};

    // pipeline prologue: fragments for k = 0
    v16bf aC  = A[aBase];
    v16bf b0C = Bc[bBase0], b1C = Bc[bBase1], b2C = Bc[bBase2], b3C = Bc[bBase3];

    for (int k = 0; k < KT_COMB - 1; ++k) {
        size_t kn = (size_t)(k + 1) * 32;
        // issue next-k loads before consuming current fragments
        v16bf aN  = A[aBase + kn];
        v16bf b0N = Bc[bBase0 + kn];
        v16bf b1N = Bc[bBase1 + kn];
        v16bf b2N = Bc[bBase2 + kn];
        v16bf b3N = Bc[bBase3 + kn];
        acc0 = WMMA_BF16(aC, b0C, acc0);
        acc1 = WMMA_BF16(aC, b1C, acc1);
        acc2 = WMMA_BF16(aC, b2C, acc2);
        acc3 = WMMA_BF16(aC, b3C, acc3);
        aC = aN; b0C = b0N; b1C = b1N; b2C = b2N; b3C = b3N;
    }
    // epilogue: last k-step
    acc0 = WMMA_BF16(aC, b0C, acc0);
    acc1 = WMMA_BF16(aC, b1C, acc1);
    acc2 = WMMA_BF16(aC, b2C, acc2);
    acc3 = WMMA_BF16(aC, b3C, acc3);

    int half = lane >> 4;
#pragma unroll
    for (int q = 0; q < 4; ++q) {
        v8f acc = (q == 0) ? acc0 : (q == 1) ? acc1 : (q == 2) ? acc2 : acc3;
        int n = (nGrp * 4 + q) * 16 + (lane & 15);
        float bias = b_ih[n] + b_hh[n];
#pragma unroll
        for (int r = 0; r < 8; ++r)
            gates[(size_t)(mTile * 16 + r + 8 * half) * G4 + n] = acc[r] + bias;
    }
}

// ---------------------------------------------------------------------------
// Per step: LSTM elementwise update; writes h,c and output slice.
// ---------------------------------------------------------------------------
__global__ void lstm_update_kernel(const float* __restrict__ gates,
                                   float* __restrict__ h, float* __restrict__ c,
                                   float* __restrict__ out, int step) {
    int idx = blockIdx.x * blockDim.x + threadIdx.x;        // 256*1024
    int b = idx >> 10, j = idx & 1023;
    const float* g = gates + (size_t)b * G4;
    float gi = g[j], gf = g[j + 1024], gg = g[j + 2048], go = g[j + 3072];
    float si = 1.f / (1.f + __expf(-gi));
    float sf = 1.f / (1.f + __expf(-gf));
    float so = 1.f / (1.f + __expf(-go));
    float cn = sf * c[idx] + si * tanhf(gg);
    float hn = so * tanhf(cn);
    c[idx] = cn;
    h[idx] = hn;
    out[((size_t)b * T_FEAT + step) * HID + j] = hn;
}

// ---------------------------------------------------------------------------
extern "C" void kernel_launch(void* const* d_in, const int* in_sizes, int n_in,
                              void* d_out, int out_size, void* d_ws, size_t ws_size,
                              hipStream_t stream) {
    const float* dh      = (const float*)d_in[0];   // (64,1,256,1024)
    const int*   caps    = (const int*)  d_in[1];   // (64,256)
    const float* W_attn  = (const float*)d_in[3];   // (128,1024)
    const float* U_attn  = (const float*)d_in[4];   // (128,1024)
    const float* b_attn  = (const float*)d_in[5];
    const float* w_attn  = (const float*)d_in[6];
    const float* W_ih    = (const float*)d_in[7];   // (4096,1024)
    const float* W_hh    = (const float*)d_in[8];   // (4096,1024)
    const float* b_ih    = (const float*)d_in[9];
    const float* b_hh    = (const float*)d_in[10];
    float* out = (float*)d_out;

    // workspace carve-up (~34 MB total)
    char* ws = (char*)d_ws;
    size_t off = 0;
    auto carve = [&](size_t bytes) { void* p = ws + off; off += (bytes + 255) & ~(size_t)255; return p; };
    unsigned short* combFrags  = (unsigned short*)carve((size_t)NT_G * KT_COMB * 32 * 16 * 2); // 16.8 MB
    unsigned short* wattnFrags = (unsigned short*)carve((size_t)NT_A * KT_H * 32 * 16 * 2);    // 256 KB
    unsigned short* uattnFrags = (unsigned short*)carve((size_t)NT_A * KT_H * 32 * 16 * 2);    // 256 KB
    unsigned short* aFrags     = (unsigned short*)carve((size_t)MT * KT_COMB * 32 * 16 * 2);   // 1 MB
    float* Uv    = (float*)carve((size_t)B_ * T_CAP * ATTN * 4);                               // 8.4 MB
    float* Wh    = (float*)carve((size_t)B_ * ATTN * 4);
    float* ctxB  = (float*)carve((size_t)B_ * DEC * 4);
    float* hB    = (float*)carve((size_t)B_ * HID * 4);
    float* cB    = (float*)carve((size_t)B_ * HID * 4);
    float* gates = (float*)carve((size_t)B_ * G4 * 4);

    // ---- one-time prep ----
    pack_comb_kernel<<<(NT_G * KT_COMB * 32) / 256, 256, 0, stream>>>(W_hh, W_ih, combFrags);
    pack_b_kernel<<<(NT_A * KT_H * 32 + 255) / 256, 256, 0, stream>>>(W_attn, wattnFrags, NT_A, KT_H, HID);
    pack_b_kernel<<<(NT_A * KT_H * 32 + 255) / 256, 256, 0, stream>>>(U_attn, uattnFrags, NT_A, KT_H, DEC);
    uv_gemm_kernel<<<(1024 * 32) / 256, 256, 0, stream>>>(dh, uattnFrags, b_attn, Uv);
    hipMemsetAsync(hB, 0, (size_t)B_ * HID * 4, stream);
    hipMemsetAsync(cB, 0, (size_t)B_ * HID * 4, stream);
    hipMemsetAsync(aFrags, 0, (size_t)MT * KT_COMB * 32 * 16 * 2, stream);

    // ---- 28 serial recurrence steps ----
    for (int step = 0; step < T_FEAT; ++step) {
        wh_gemm_kernel<<<(MT * NT_A * 32) / 256, 256, 0, stream>>>(aFrags, wattnFrags, Wh);
        attention_ctx_kernel<<<B_, 256, 0, stream>>>(Wh, Uv, w_attn, caps, dh, ctxB);
        pack_a_kernel<<<(MT * KT_H * 32 + 255) / 256, 256, 0, stream>>>(ctxB, aFrags, KT_H);
        gates_gemm_kernel<<<(MT * 64 * 32) / 256, 256, 0, stream>>>(aFrags, combFrags, b_ih, b_hh, gates);
        lstm_update_kernel<<<(B_ * HID) / 256, 256, 0, stream>>>(gates, hB, cB, out, step);
        pack_a_kernel<<<(MT * KT_H * 32 + 255) / 256, 256, 0, stream>>>(hB, aFrags, 0);
    }
}
